// Generator_43207370998163
// MI455X (gfx1250) — matrix operand order repaired and checked by reading, not compiled, not run on hardware
//
#include <hip/hip_runtime.h>
#include <hip/hip_bf16.h>
#include <stdint.h>

#define BATCH 256
#define SEQ   256
#define EMB   128
#define HID   512
#define VOCAB 32000

typedef __attribute__((ext_vector_type(16))) __bf16 v16bf;
typedef __attribute__((ext_vector_type(8)))  float  v8f;

union Frag16 { v16bf v; uint4 q[2]; };

// ---------------- WMMA fragment helpers (contiguous k = 16h+s in A and B) ----------------
// A (16x32 bf16): lane l -> row M=l%16; elems 0..15 = K[k0+16h .. k0+16h+15],
// h = l>>4. Contiguous 16B loads.
__device__ inline void load_a_frag(Frag16& a, const __hip_bfloat16* A, int lda,
                                   int mBase, int k0, int lane) {
  int row = mBase + (lane & 15);
  const __hip_bfloat16* p = A + (size_t)row * lda + k0 + ((lane >> 4) << 4);
  a.q[0] = *(const uint4*)(p);
  a.q[1] = *(const uint4*)(p + 8);
}

// B (32x16 bf16) pre-packed lane-major: 32 lanes x 16 contiguous bf16 per tile.
__device__ inline void load_b_frag(Frag16& b, const __hip_bfloat16* Bp,
                                   int ntiles, int kt, int nt, int lane) {
  const __hip_bfloat16* p = Bp + ((((size_t)kt * ntiles + nt) * 32) + lane) * 16;
  b.q[0] = *(const uint4*)(p);
  b.q[1] = *(const uint4*)(p + 8);
}

__device__ inline v8f wmma_bf16(const Frag16& a, const Frag16& b, v8f c) {
  return __builtin_amdgcn_wmma_f32_16x16x32_bf16(false, a.v, false, b.v,
                                                 (short)0, c, false, false);
}

// ---------------- Prep: pack fp32 weight [K][N] -> bf16 WMMA-B tile layout ----
__global__ void pack_b_kernel(const float* __restrict__ src,
                              __hip_bfloat16* __restrict__ dst, int K, int N) {
  int idx = blockIdx.x * blockDim.x + threadIdx.x;
  if (idx >= K * N) return;
  int k = idx / N, n = idx % N;
  int ktile = k >> 5, kin = k & 31;
  int ntile = n >> 4, nin = n & 15;
  int lane = nin + ((kin >> 4) << 4);  // B[lane][e] = W[e + 16*(lane>=16)][lane%16]
  int e = kin & 15;
  size_t out = ((((size_t)ktile * (N >> 4) + ntile) * 32) + lane) * 16 + e;
  dst[out] = __float2bfloat16(src[idx]);
}

__global__ void convert_bf16_kernel(const float* __restrict__ src,
                                    __hip_bfloat16* __restrict__ dst, int n) {
  int idx = blockIdx.x * blockDim.x + threadIdx.x;
  if (idx < n) dst[idx] = __float2bfloat16(src[idx]);
}

__global__ void init_state_kernel(float* __restrict__ c,
                                  __hip_bfloat16* __restrict__ hbf,
                                  __hip_bfloat16* __restrict__ xt,
                                  const __hip_bfloat16* __restrict__ embb,
                                  const int* __restrict__ start_tok) {
  int idx = blockIdx.x * blockDim.x + threadIdx.x;
  if (idx < BATCH * HID) { c[idx] = 0.0f; hbf[idx] = __float2bfloat16(0.0f); }
  if (idx < BATCH * EMB) {
    int st = *start_tok;
    xt[idx] = embb[(size_t)st * EMB + (idx & (EMB - 1))];
  }
}

// ---------------- Gates: Z[g] = xt@Wg + h@Ug + bg (bf16 WMMA, fp32 acc) -------
// 4 gates * 16 Mtiles * 8 Ngroups = 512 wave-tiles = 128 blocks * 4 waves (no divergence)
__global__ __launch_bounds__(128)
void gates_gemm_kernel(const __hip_bfloat16* __restrict__ xt,
                       const __hip_bfloat16* __restrict__ h,
                       const __hip_bfloat16* __restrict__ Wp,
                       const __hip_bfloat16* __restrict__ Up,
                       const float* __restrict__ bi, const float* __restrict__ bf_,
                       const float* __restrict__ bog, const float* __restrict__ bc,
                       float* __restrict__ Z) {
  const int lane = threadIdx.x & 31;
  const int wid = blockIdx.x * 4 + (threadIdx.x >> 5);
  const int gate = wid >> 7;
  const int rem = wid & 127;
  const int mBase = (rem >> 3) * 16;
  const int nGroup = rem & 7;

  v8f acc[4];
#pragma unroll
  for (int j = 0; j < 4; ++j)
#pragma unroll
    for (int e = 0; e < 8; ++e) acc[j][e] = 0.0f;

  // phase 1: xt(256x128) @ W(128x512)
  const __hip_bfloat16* Wg = Wp + (size_t)gate * EMB * HID;
#pragma unroll
  for (int kt = 0; kt < EMB / 32; ++kt) {
    Frag16 a; load_a_frag(a, xt, EMB, mBase, kt * 32, lane);
#pragma unroll
    for (int j = 0; j < 4; ++j) {
      Frag16 b; load_b_frag(b, Wg, HID / 16, kt, nGroup * 4 + j, lane);
      acc[j] = wmma_bf16(a, b, acc[j]);
    }
  }
  // phase 2: h(256x512) @ U(512x512)
  const __hip_bfloat16* Ug = Up + (size_t)gate * HID * HID;
  for (int kt = 0; kt < HID / 32; ++kt) {
    Frag16 a; load_a_frag(a, h, HID, mBase, kt * 32, lane);
#pragma unroll
    for (int j = 0; j < 4; ++j) {
      Frag16 b; load_b_frag(b, Ug, HID / 16, kt, nGroup * 4 + j, lane);
      acc[j] = wmma_bf16(a, b, acc[j]);
    }
  }

  const float* bias = (gate == 0) ? bi : (gate == 1) ? bf_ : (gate == 2) ? bog : bc;
  float* Zg = Z + (size_t)gate * BATCH * HID;
  const int m0 = mBase + ((lane >> 4) << 3);
  const int c0 = nGroup * 64 + (lane & 15);
#pragma unroll
  for (int j = 0; j < 4; ++j)
#pragma unroll
    for (int vv = 0; vv < 8; ++vv)
      Zg[(size_t)(m0 + vv) * HID + c0 + j * 16] = acc[j][vv] + bias[c0 + j * 16];
}

// ---------------- Pointwise LSTM cell update ---------------------------------
__device__ inline float sigmoidf_(float x) { return 1.0f / (1.0f + __expf(-x)); }

__global__ void lstm_pointwise_kernel(const float* __restrict__ Z,
                                      float* __restrict__ c,
                                      __hip_bfloat16* __restrict__ hbf) {
  int idx = blockIdx.x * blockDim.x + threadIdx.x;
  if (idx >= BATCH * HID) return;
  const int N = BATCH * HID;
  float ig = sigmoidf_(Z[idx]);
  float fg = sigmoidf_(Z[idx + N]);
  float og = sigmoidf_(Z[idx + 2 * N]);
  float gg = tanhf(Z[idx + 3 * N]);
  float cn = fg * c[idx] + ig * gg;
  float hn = og * tanhf(cn);
  c[idx] = cn;
  hbf[idx] = __float2bfloat16(hn);
}

// ---------------- Logits: h(256x512) @ Wo(512x32000) + bo --------------------
// 16 Mtiles * 500 Ngroups = 8000 wave-tiles = 2000 blocks * 4 waves (no divergence)
__global__ __launch_bounds__(128)
void logits_gemm_kernel(const __hip_bfloat16* __restrict__ h,
                        const __hip_bfloat16* __restrict__ WoP,
                        const float* __restrict__ bo,
                        float* __restrict__ logits) {
  const int lane = threadIdx.x & 31;
  const int wid = blockIdx.x * 4 + (threadIdx.x >> 5);
  const int mBase = (wid / 500) * 16;
  const int nGroup = wid % 500;
  const int NT = VOCAB / 16;  // 2000

  v8f acc[4];
#pragma unroll
  for (int j = 0; j < 4; ++j)
#pragma unroll
    for (int e = 0; e < 8; ++e) acc[j][e] = 0.0f;

  for (int kt = 0; kt < HID / 32; ++kt) {
    Frag16 a; load_a_frag(a, h, HID, mBase, kt * 32, lane);
    if (kt + 1 < HID / 32) {  // prefetch next K-tile of B (global_prefetch_b8)
      const __hip_bfloat16* pf =
          WoP + ((((size_t)(kt + 1) * NT + nGroup * 4) * 32) + lane) * 16;
      __builtin_prefetch(pf, 0, 3);
    }
#pragma unroll
    for (int j = 0; j < 4; ++j) {
      Frag16 b; load_b_frag(b, WoP, NT, kt, nGroup * 4 + j, lane);
      acc[j] = wmma_bf16(a, b, acc[j]);
    }
  }

  const int m0 = mBase + ((lane >> 4) << 3);
  const int c0 = nGroup * 64 + (lane & 15);
#pragma unroll
  for (int j = 0; j < 4; ++j)
#pragma unroll
    for (int vv = 0; vv < 8; ++vv)
      logits[(size_t)(m0 + vv) * VOCAB + c0 + j * 16] = acc[j][vv] + bo[c0 + j * 16];
}

// ---------------- Softmax + categorical sample + embed gather ----------------
__device__ inline float uniform01_(unsigned t, unsigned b) {
  unsigned x = t * 9781u + b * 6271u + 0x9E3779B9u;
  x ^= x >> 17; x *= 0xed5ad4bbu;
  x ^= x >> 11; x *= 0xac4c1b51u;
  x ^= x >> 15; x *= 0x31848babu;
  x ^= x >> 14;
  return ((float)x + 0.5f) * (1.0f / 4294967296.0f);
}

__global__ __launch_bounds__(256)
void sample_kernel(const float* __restrict__ logits,
                   const __hip_bfloat16* __restrict__ embb,
                   __hip_bfloat16* __restrict__ xt,
                   float* __restrict__ out_tok,   // [BATCH][SEQ]
                   float* __restrict__ out_prob,  // [SEQ][BATCH]
                   int t) {
  const int b = blockIdx.x;
  const int tid = threadIdx.x;
  const int CHUNK = VOCAB / 256;  // 125, contiguous per thread for exact CDF
  const float* row = logits + (size_t)b * VOCAB;

  __shared__ float sh[256];
  __shared__ float shp[257];
  __shared__ int stok;
  __shared__ float sprob;

  // 1) row max
  float m = -1e30f;
  for (int i = tid; i < VOCAB; i += 256) m = fmaxf(m, row[i]);
  sh[tid] = m; __syncthreads();
  for (int s = 128; s > 0; s >>= 1) {
    if (tid < s) sh[tid] = fmaxf(sh[tid], sh[tid + s]);
    __syncthreads();
  }
  float mx = sh[0]; __syncthreads();

  // 2) per-thread contiguous chunk sum of exp
  int c0 = tid * CHUNK, c1 = c0 + CHUNK;
  float s = 0.0f;
  for (int i = c0; i < c1; ++i) s += __expf(row[i] - mx);
  sh[tid] = s; __syncthreads();

  // 3) exclusive prefix (serial: exact partition of [0,Z))
  if (tid == 0) {
    float a = 0.0f;
    for (int i = 0; i < 256; ++i) { shp[i] = a; a += sh[i]; }
    shp[256] = a;
  }
  __syncthreads();
  float Zt = shp[256];
  float u = uniform01_((unsigned)t, (unsigned)b) * Zt;

  // 4) exactly one thread owns the interval containing u
  if (u >= shp[tid] && (tid == 255 || u < shp[tid + 1])) {
    float a = shp[tid];
    int tok = c1 - 1;
    for (int i = c0; i < c1; ++i) {
      a += __expf(row[i] - mx);
      if (u < a) { tok = i; break; }
    }
    stok = tok;
    sprob = __expf(row[tok] - mx) / Zt;
  }
  __syncthreads();

  if (tid == 0) {
    out_tok[(size_t)b * SEQ + t] = (float)stok;   // gen_x.T -> [B,S]
    out_prob[(size_t)t * BATCH + b] = sprob;      // gen_o   -> [S,B]
  }
  // 5) fused embedding gather for next step
  for (int i = tid; i < EMB; i += 256)
    xt[(size_t)b * EMB + i] = embb[(size_t)stok * EMB + i];
}

// ---------------- Host launcher ----------------------------------------------
extern "C" void kernel_launch(void* const* d_in, const int* in_sizes, int n_in,
                              void* d_out, int out_size, void* d_ws, size_t ws_size,
                              hipStream_t stream) {
  const float* emb = (const float*)d_in[2];
  const float* Wi  = (const float*)d_in[3];
  const float* Ui  = (const float*)d_in[4];
  const float* bi  = (const float*)d_in[5];
  const float* Wf  = (const float*)d_in[6];
  const float* Uf  = (const float*)d_in[7];
  const float* bf_ = (const float*)d_in[8];
  const float* Wog = (const float*)d_in[9];
  const float* Uog = (const float*)d_in[10];
  const float* bog = (const float*)d_in[11];
  const float* Wc  = (const float*)d_in[12];
  const float* Uc  = (const float*)d_in[13];
  const float* bc  = (const float*)d_in[14];
  const float* Wo  = (const float*)d_in[15];
  const float* bo  = (const float*)d_in[16];
  const int* start = (const int*)d_in[17];

  // workspace layout (all 256B aligned)
  size_t off = 0;
  const size_t EMB_BF = off; off += (size_t)VOCAB * EMB * 2;        // 8.19 MB
  const size_t WG_BF  = off; off += (size_t)4 * EMB * HID * 2;      // 0.52 MB
  const size_t UG_BF  = off; off += (size_t)4 * HID * HID * 2;      // 2.10 MB
  const size_t WO_BF  = off; off += (size_t)HID * VOCAB * 2;        // 32.77 MB
  const size_t H_BF   = off; off += (size_t)BATCH * HID * 2;
  const size_t C_F    = off; off += (size_t)BATCH * HID * 4;
  const size_t XT_BF  = off; off += (size_t)BATCH * EMB * 2;
  const size_t Z_F    = off; off += (size_t)4 * BATCH * HID * 4;
  const size_t LOG_F  = off; off += (size_t)BATCH * VOCAB * 4;      // 32.77 MB
  if (ws_size < off) return;

  char* ws = (char*)d_ws;
  __hip_bfloat16* embb = (__hip_bfloat16*)(ws + EMB_BF);
  __hip_bfloat16* WGp  = (__hip_bfloat16*)(ws + WG_BF);
  __hip_bfloat16* UGp  = (__hip_bfloat16*)(ws + UG_BF);
  __hip_bfloat16* WOp  = (__hip_bfloat16*)(ws + WO_BF);
  __hip_bfloat16* hbf  = (__hip_bfloat16*)(ws + H_BF);
  float*          cst  = (float*)(ws + C_F);
  __hip_bfloat16* xt   = (__hip_bfloat16*)(ws + XT_BF);
  float*          Zbuf = (float*)(ws + Z_F);
  float*          lg   = (float*)(ws + LOG_F);

  float* out_tok  = (float*)d_out;
  float* out_prob = (float*)d_out + (size_t)BATCH * SEQ;

  // --- one-time prep (per launch): convert + pack weights into L2-resident bf16
  {
    int n, blk = 256;
    n = VOCAB * EMB;
    convert_bf16_kernel<<<(n + blk - 1) / blk, blk, 0, stream>>>(emb, embb, n);
    n = EMB * HID;
    pack_b_kernel<<<(n + blk - 1) / blk, blk, 0, stream>>>(Wi,  WGp + 0 * (size_t)n, EMB, HID);
    pack_b_kernel<<<(n + blk - 1) / blk, blk, 0, stream>>>(Wf,  WGp + 1 * (size_t)n, EMB, HID);
    pack_b_kernel<<<(n + blk - 1) / blk, blk, 0, stream>>>(Wog, WGp + 2 * (size_t)n, EMB, HID);
    pack_b_kernel<<<(n + blk - 1) / blk, blk, 0, stream>>>(Wc,  WGp + 3 * (size_t)n, EMB, HID);
    n = HID * HID;
    pack_b_kernel<<<(n + blk - 1) / blk, blk, 0, stream>>>(Ui,  UGp + 0 * (size_t)n, HID, HID);
    pack_b_kernel<<<(n + blk - 1) / blk, blk, 0, stream>>>(Uf,  UGp + 1 * (size_t)n, HID, HID);
    pack_b_kernel<<<(n + blk - 1) / blk, blk, 0, stream>>>(Uog, UGp + 2 * (size_t)n, HID, HID);
    pack_b_kernel<<<(n + blk - 1) / blk, blk, 0, stream>>>(Uc,  UGp + 3 * (size_t)n, HID, HID);
    n = HID * VOCAB;
    pack_b_kernel<<<(n + blk - 1) / blk, blk, 0, stream>>>(Wo, WOp, HID, VOCAB);
    init_state_kernel<<<(BATCH * HID + 255) / 256, 256, 0, stream>>>(cst, hbf, xt, embb, start);
  }

  // --- sequential generation: 256 steps, 4 kernels each, all stream-ordered
  for (int t = 0; t < SEQ; ++t) {
    gates_gemm_kernel<<<128, 128, 0, stream>>>(xt, hbf, WGp, UGp, bi, bf_, bog, bc, Zbuf);
    lstm_pointwise_kernel<<<(BATCH * HID + 255) / 256, 256, 0, stream>>>(Zbuf, cst, hbf);
    logits_gemm_kernel<<<2000, 128, 0, stream>>>(hbf, WOp, bo, lg);
    sample_kernel<<<BATCH, 256, 0, stream>>>(lg, embb, xt, out_tok, out_prob, t);
  }
}
